// QuantLinear_1726576854707
// MI455X (gfx1250) — matrix operand order repaired and checked by reading, not compiled, not run on hardware
//
#include <hip/hip_runtime.h>
#include <hip/hip_fp16.h>
#include <stdint.h>

typedef __attribute__((ext_vector_type(16))) _Float16 v16h;
typedef __attribute__((ext_vector_type(8)))  float    v8f;
typedef __attribute__((ext_vector_type(2)))  _Float16 h2;
typedef __attribute__((ext_vector_type(4)))  uint32_t u32x4;
typedef __attribute__((ext_vector_type(8)))  int      i32x8;
typedef __attribute__((ext_vector_type(4)))  int      i32x4;

#define BM 128
#define BN 128
#define BK 64
#define LDSW 72            // halfs per LDS row: 64 + 8 pad (144B, 16B aligned)
#define K_DIM 4096
#define N_DIM 4096

#if __has_builtin(__builtin_amdgcn_tensor_load_to_lds)
#define USE_TDM 1
#else
#define USE_TDM 0
#endif

union W8    { uint4 u; h2 p[4]; };
union AFrag { v16h v; uint4 u[2]; };
union C2    { uint32_t u; h2 h; };

__global__ __launch_bounds__(256) void gptq_wmma_kernel(
    const _Float16* __restrict__ x,
    const uint32_t* __restrict__ qweight,
    const uint32_t* __restrict__ qzeros,
    const _Float16* __restrict__ scales,
    const _Float16* __restrict__ bias,
    float* __restrict__ out,
    int M)
{
    __shared__ __attribute__((aligned(16))) _Float16 lds_a[2][BM * LDSW];
    __shared__ __attribute__((aligned(16))) _Float16 lds_w[BN * LDSW];

    const int tid  = threadIdx.x;
    const int lane = tid & 31;
    const int block_m = blockIdx.y * BM;
    const int block_n = blockIdx.x * BN;

    const int wv = __builtin_amdgcn_readfirstlane(tid >> 5);  // wave id (SGPR)
    const int wm = (wv & 3) * 32;   // wave's row offset inside C tile
    const int wn = (wv >> 2) * 64;  // wave's col offset inside C tile

    v8f acc[2][4] = {};

    // ---- register staging for the quantized-weight pipeline ----
    uint32_t qreg[4];
    _Float16 sreg[4];   // scale
    _Float16 zreg[4];   // -(1025 + z)*scale  (folds the +1 and the 0x6400 bias)

    // ================= A-tile staging: Tensor Data Mover =================
#if USE_TDM
    // Each wave DMAs its 16-row slice of the 128x64 A tile; HW pads each
    // 64-half line by 4 dwords to produce the 72-half LDS stride.
    const uint32_t lds_a_base = (uint32_t)(size_t)(&lds_a[0][0]);
    auto copy_a_tile = [&](int kbase, int buf) {
        uint64_t ga = (uint64_t)(size_t)x +
                      (((uint64_t)(block_m + wv * 16)) * K_DIM + (uint64_t)kbase) * 2u;
        u32x4 g0;
        g0.x = 1u;                                   // count=1 (valid), user mode
        g0.y = lds_a_base + (uint32_t)(buf * BM * LDSW + wv * 16 * LDSW) * 2u;
        g0.z = (uint32_t)ga;                         // global_addr[31:0]
        g0.w = (uint32_t)(ga >> 32) | (2u << 30);    // global_addr[56:32], type=2
        i32x8 g1;
        g1[0] = (int)((1u << 16)      // data_size = 2 bytes
                    | (1u << 20)      // pad_enable
                    | (4u << 22)      // pad_interval: 32 dwords (one 64-half line)
                    | (3u << 25));    // pad_amount: 4 dwords (8 halfs)
        g1[1] = (int)(((uint32_t)K_DIM & 0xFFFFu) << 16);            // tensor_dim0 lo16
        g1[2] = (int)(((uint32_t)K_DIM >> 16) |
                      (((uint32_t)M & 0xFFFFu) << 16));              // dim0 hi / dim1 lo
        g1[3] = (int)((((uint32_t)M >> 16) & 0xFFFFu) | (64u << 16));// dim1 hi / tile_dim0=64
        g1[4] = 16;                                                  // tile_dim1=16
        g1[5] = K_DIM;                                               // tensor_dim0_stride lo32
        g1[6] = 0;
        g1[7] = 0;
        i32x4 z4 = {0, 0, 0, 0};
#if defined(__clang_major__) && (__clang_major__ >= 23)
        i32x8 z8 = {0, 0, 0, 0, 0, 0, 0, 0};
        __builtin_amdgcn_tensor_load_to_lds(g0, g1, z4, z4, z8, 0);
#else
        __builtin_amdgcn_tensor_load_to_lds(g0, g1, z4, z4, 0);
#endif
    };
    auto wait_a_tile = [&]() {
#if __has_builtin(__builtin_amdgcn_s_wait_tensorcnt)
        __builtin_amdgcn_s_wait_tensorcnt(0);
#else
        asm volatile("s_wait_tensorcnt 0x0" ::: "memory");
#endif
    };
#else
    // Fallback: cooperative 256-thread copy (4x b128 each)
    auto copy_a_tile = [&](int kbase, int buf) {
        #pragma unroll
        for (int i = 0; i < 4; ++i) {
            int u = tid + i * 256;
            int row = u >> 3, chunk = u & 7;
            uint4 v = *(const uint4*)(x + (size_t)(block_m + row) * K_DIM
                                        + kbase + chunk * 8);
            *(uint4*)(&lds_a[buf][row * LDSW + chunk * 8]) = v;
        }
    };
    auto wait_a_tile = [&]() {};
#endif

    // ================= quantized W: global load stage =================
    auto load_w = [&](int kbase) {
        #pragma unroll
        for (int i = 0; i < 4; ++i) {
            int u = tid + i * 256;                   // 8 k-rows x 128 cols
            int krow = u >> 7, nl = u & 127;
            int kk = kbase + krow * 8;
            int g  = kk >> 7;                        // group = k/128 (== g_idx[k])
            int n  = block_n + nl;
            qreg[i] = qweight[(size_t)(kk >> 3) * N_DIM + n];
            uint32_t zq = qzeros[(size_t)g * (N_DIM / 8) + (n >> 3)];
            uint32_t z  = (zq >> ((n & 7) * 4)) & 15u;
            _Float16 s  = scales[(size_t)g * N_DIM + n];
            sreg[i] = s;
            zreg[i] = -((_Float16)(int)(z + 1025u) * s);  // -(1024 + z + 1)*s
        }
    };

    // ======== packed dequant + commit W tile to LDS ([n][k] layout) ========
    auto store_w = [&]() {
        #pragma unroll
        for (int i = 0; i < 4; ++i) {
            int u = tid + i * 256;
            int krow = u >> 7, nl = u & 127;
            uint32_t q = qreg[i];
            h2 s2  = {sreg[i], sreg[i]};
            h2 zs2 = {zreg[i], zreg[i]};
            // nibble pairs -> (1024+w) packed fp16 via 0x6400 mantissa trick
            uint32_t b0 = 0x64006400u | (q & 0xFu)          | ((q << 12) & 0x000F0000u);
            uint32_t b1 = 0x64006400u | ((q >>  8) & 0xFu)  | ((q <<  4) & 0x000F0000u);
            uint32_t b2 = 0x64006400u | ((q >> 16) & 0xFu)  | ((q >>  4) & 0x000F0000u);
            uint32_t b3 = 0x64006400u | ((q >> 24) & 0xFu)  | ((q >> 12) & 0x000F0000u);
            W8 w;
            C2 c;
            c.u = b0; w.p[0] = c.h * s2 + zs2;   // v_pk_fma_f16
            c.u = b1; w.p[1] = c.h * s2 + zs2;
            c.u = b2; w.p[2] = c.h * s2 + zs2;
            c.u = b3; w.p[3] = c.h * s2 + zs2;
            *(uint4*)(&lds_w[nl * LDSW + krow * 8]) = w.u;
        }
    };

    // ======== WMMA over one staged 64-deep k-tile (2 steps of K=32) ========
    auto compute_tile = [&](int buf) {
        const _Float16* A = lds_a[buf];
        #pragma unroll
        for (int ks = 0; ks < 2; ++ks) {
            AFrag a[2], b[4];
            #pragma unroll
            for (int im = 0; im < 2; ++im) {
                // A (16-bit, 16x32): lanes0-15 K=0..7 & 16..23; lanes16-31 +8
                int row = wm + im * 16 + (lane & 15);
                int kb  = ks * 32 + ((lane >> 4) << 3);
                const uint4* p = (const uint4*)(A + row * LDSW + kb);
                a[im].u[0] = p[0];
                a[im].u[1] = p[2];
            }
            #pragma unroll
            for (int jn = 0; jn < 4; ++jn) {
                // B (16-bit, 32x16): lanes0-15 K=0..7 & 16..23; lanes16-31 +8
                int n  = wn + jn * 16 + (lane & 15);
                int kb = ks * 32 + ((lane >> 4) << 3);
                const uint4* p = (const uint4*)(lds_w + n * LDSW + kb);
                b[jn].u[0] = p[0];
                b[jn].u[1] = p[2];
            }
            #pragma unroll
            for (int im = 0; im < 2; ++im)
                #pragma unroll
                for (int jn = 0; jn < 4; ++jn)
                    acc[im][jn] = __builtin_amdgcn_wmma_f32_16x16x32_f16(
                        false, a[im].v, false, b[jn].v,
                        (short)0, acc[im][jn], false, false);
        }
    };

    // ================= software-pipelined main loop =================
    copy_a_tile(0, 0);
    load_w(0);
    store_w();
    wait_a_tile();
    __syncthreads();

    for (int kt = 1; kt < K_DIM / BK; ++kt) {
        copy_a_tile(kt * BK, kt & 1);   // async DMA next A tile (other buffer)
        load_w(kt * BK);                // prefetch next quantized W into regs
        compute_tile((kt - 1) & 1);     // WMMA on current tile
        __syncthreads();                // everyone done reading lds_w
        store_w();                      // publish dequantized W(t)
        wait_a_tile();                  // my TDM slice landed
        __syncthreads();                // tile t fully visible
    }
    compute_tile((K_DIM / BK - 1) & 1);

    // ================= epilogue: fp32 acc + bias =================
    #pragma unroll
    for (int im = 0; im < 2; ++im) {
        #pragma unroll
        for (int jn = 0; jn < 4; ++jn) {
            int col = block_n + wn + jn * 16 + (lane & 15);
            float bv = (float)bias[col];
            #pragma unroll
            for (int r = 0; r < 8; ++r) {
                int row = block_m + wm + im * 16 + r + ((lane >> 4) << 3);
                out[(size_t)row * N_DIM + col] = acc[im][jn][r] + bv;
            }
        }
    }
}

extern "C" void kernel_launch(void* const* d_in, const int* in_sizes, int n_in,
                              void* d_out, int out_size, void* d_ws, size_t ws_size,
                              hipStream_t stream) {
    const _Float16* x       = (const _Float16*)d_in[0];
    const uint32_t* qweight = (const uint32_t*)d_in[1];
    const uint32_t* qzeros  = (const uint32_t*)d_in[2];
    const _Float16* scales  = (const _Float16*)d_in[3];
    // d_in[4] = g_idx; for GROUP_SIZE=128 it is exactly k>>7, computed in-kernel
    const _Float16* bias    = (const _Float16*)d_in[5];
    float* out = (float*)d_out;

    int M = in_sizes[0] / K_DIM;        // 4*2048 = 8192
    dim3 grid(N_DIM / BN, M / BM);      // 32 x 64 blocks
    gptq_wmma_kernel<<<grid, 256, 0, stream>>>(x, qweight, qzeros, scales, bias, out, M);
}
